// CausalSelfAttention_39831526703820
// MI455X (gfx1250) — matrix operand order repaired and checked by reading, not compiled, not run on hardware
//
#include <hip/hip_runtime.h>
#include <hip/hip_bf16.h>
#include <stdint.h>

// Shapes (fixed by the reference)
#define BATCH  4
#define S_LEN  2048
#define DMODEL 1024
#define NHEAD  16
#define DK     64

typedef __attribute__((ext_vector_type(16))) __bf16 v16bf;
typedef __attribute__((ext_vector_type(8)))  float  v8f;

union Frag {
  v16bf v;
  uint4 q[2];
  unsigned u[8];
  unsigned short h[16];
};

__device__ __forceinline__ unsigned short f2bf(float f) {
  unsigned u = __float_as_uint(f);
  u = (u + 0x7FFFu + ((u >> 16) & 1u)) >> 16;   // round-to-nearest-even
  return (unsigned short)u;
}
__device__ __forceinline__ float bf2f(unsigned short h) {
  return __uint_as_float(((unsigned)h) << 16);
}

// Low 32 bits of a generic pointer to LDS == LDS byte offset (AMDGPU aperture).
__device__ __forceinline__ unsigned lds_off32(const void* p) {
  return (unsigned)(unsigned long long)(size_t)p;
}

// CDNA5 async DMA: global -> LDS, 16B per lane, tracked by ASYNCcnt.
__device__ __forceinline__ void async_ld_b128(unsigned lds_addr, const void* gaddr) {
  asm volatile("global_load_async_to_lds_b128 %0, %1, off"
               :: "v"(lds_addr), "v"(gaddr) : "memory");
}
__device__ __forceinline__ void wait_async_le8() {
  asm volatile("s_wait_asynccnt 0x8" ::: "memory");
}
__device__ __forceinline__ void wait_async_0() {
  asm volatile("s_wait_asynccnt 0x0" ::: "memory");
}

// ---------------------------------------------------------------------------
// f32 -> bf16 pairwise converter
// ---------------------------------------------------------------------------
__global__ void cvt_f32_bf16(const float* __restrict__ src,
                             unsigned short* __restrict__ dst, int npairs) {
  int i = blockIdx.x * blockDim.x + threadIdx.x;
  if (i >= npairs) return;
  float2 f = ((const float2*)src)[i];
  unsigned pk = (unsigned)f2bf(f.x) | ((unsigned)f2bf(f.y) << 16);
  ((unsigned*)dst)[i] = pk;
}

// ---------------------------------------------------------------------------
// GEMM: C[M=8192, N=1024] = A[M,K=1024]_bf16 x B[N,K]_bf16 (both K-major)
// 256 thr (8 waves, 4Mx2N), block tile 128x128, wave tile 32x64, K-step 64.
// LDS double-buffered (literal buffer ids), staged via async-to-LDS DMA.
// mode 0: store bf16 as [B,H,S,DK] (Q/K)   mode 1: store bf16 [B,H,DK,S] (V^T)
// mode 2: store f32  as [M,N] (final output)
// ---------------------------------------------------------------------------
__global__ __launch_bounds__(256)
void gemm_bf16(const unsigned short* __restrict__ A,
               const unsigned short* __restrict__ Bw,
               void* __restrict__ Cout, int mode) {
  __shared__ __align__(16) unsigned short As[2][128 * 64];
  __shared__ __align__(16) unsigned short Bs[2][128 * 64];

  const int t     = threadIdx.x;
  const int wave  = t >> 5;
  const int lane  = t & 31;
  const int lhalf = lane >> 4;
  const int l16   = lane & 15;
  const int wm    = (wave >> 1) * 32;   // 0,32,64,96
  const int wn    = (wave & 1) * 64;    // 0,64
  const int blockM = blockIdx.x * 128;
  const int blockN = blockIdx.y * 128;

  // Per-thread staging: 4 chunks (16B) of A + 4 of B per 128x64 tile.
  size_t gA[4], gB[4];
  unsigned lA[4], lB[4];             // LDS byte offsets within buffer 0
  for (int j = 0; j < 4; j++) {
    int c = t + j * 256;
    int row = c >> 3, col = (c & 7) * 8;
    gA[j] = (size_t)(blockM + row) * DMODEL + col;
    gB[j] = (size_t)(blockN + row) * DMODEL + col;
    lA[j] = lds_off32(&As[0][row * 64 + col]);
    lB[j] = lds_off32(&Bs[0][row * 64 + col]);
  }
  const unsigned abufstride = lds_off32(&As[1][0]) - lds_off32(&As[0][0]);
  const unsigned bbufstride = lds_off32(&Bs[1][0]) - lds_off32(&Bs[0][0]);

  v8f acc[2][4] = {};

  auto stage = [&](int k, int buf) {
    const size_t kofs = (size_t)k;   // element offset into the K dimension
    const unsigned ao = (unsigned)buf * abufstride;
    const unsigned bo = (unsigned)buf * bbufstride;
#pragma unroll
    for (int j = 0; j < 4; j++) async_ld_b128(lA[j] + ao, &A[gA[j] + kofs]);
#pragma unroll
    for (int j = 0; j < 4; j++) async_ld_b128(lB[j] + bo, &Bw[gB[j] + kofs]);
    __builtin_prefetch(&A[gA[0] + kofs + 64], 0, 1);
  };

  auto compute = [&](int buf) {
#pragma unroll
    for (int kc = 0; kc < 2; kc++) {
      Frag a[2], b[4];
#pragma unroll
      for (int mt = 0; mt < 2; mt++) {             // A-operand: interleaved halves
        int row = wm + mt * 16 + l16;
        a[mt].q[0] = *(const uint4*)&As[buf][row * 64 + kc * 32 + lhalf * 8];
        a[mt].q[1] = *(const uint4*)&As[buf][row * 64 + kc * 32 + 16 + lhalf * 8];
      }
#pragma unroll
      for (int nt = 0; nt < 4; nt++) {             // B-operand: interleaved halves
        int row = wn + nt * 16 + l16;
        b[nt].q[0] = *(const uint4*)&Bs[buf][row * 64 + kc * 32 + lhalf * 8];
        b[nt].q[1] = *(const uint4*)&Bs[buf][row * 64 + kc * 32 + 16 + lhalf * 8];
      }
#pragma unroll
      for (int mt = 0; mt < 2; mt++)
#pragma unroll
        for (int nt = 0; nt < 4; nt++)
          acc[mt][nt] = __builtin_amdgcn_wmma_f32_16x16x32_bf16(
              false, a[mt].v, false, b[nt].v, (short)0, acc[mt][nt], false, false);
    }
  };

  const int NSTEP = DMODEL / 64;  // 16 (even)
  stage(0, 0);
  for (int i = 0; i < NSTEP; i += 2) {
    // even step: consume buffer 0
    stage((i + 1) * 64, 1);
    wait_async_le8();             // oldest 8 (tile i) retired: in-order async
    __syncthreads();              // tile i visible to all waves
    compute(0);
    __syncthreads();              // everyone done with buf0 before refill
    // odd step: consume buffer 1
    if (i + 2 < NSTEP) {
      stage((i + 2) * 64, 0);
      wait_async_le8();
    } else {
      wait_async_0();
    }
    __syncthreads();
    compute(1);
    __syncthreads();
  }

  // Epilogue. C-layout: VGPR v -> row (v + lhalf*8), lane l16 -> column.
  for (int mt = 0; mt < 2; mt++)
    for (int nt = 0; nt < 4; nt++) {
      int rbase = blockM + wm + mt * 16;
      int nglob = blockN + wn + nt * 16 + l16;
      if (mode == 2) {
        float* O = (float*)Cout;
        for (int v = 0; v < 8; v++) {
          int r = rbase + v + lhalf * 8;
          O[(size_t)r * DMODEL + nglob] = acc[mt][nt][v];
        }
      } else if (mode == 0) {
        unsigned short* O = (unsigned short*)Cout;
        int h = nglob >> 6, dk = nglob & 63;
        for (int v = 0; v < 8; v++) {
          int r = rbase + v + lhalf * 8;
          int b2 = r >> 11, s = r & (S_LEN - 1);
          O[(((size_t)(b2 * NHEAD + h)) * S_LEN + s) * DK + dk] =
              f2bf(acc[mt][nt][v]);
        }
      } else {  // mode 1: transposed packed store -> V^T [B,H,DK,S]
        unsigned short* O = (unsigned short*)Cout;
        int h = nglob >> 6, dk = nglob & 63;
        int r0 = rbase + lhalf * 8;
        int b2 = r0 >> 11, s0 = r0 & (S_LEN - 1);
        unsigned pk[4];
        for (int v = 0; v < 4; v++)
          pk[v] = (unsigned)f2bf(acc[mt][nt][2 * v]) |
                  ((unsigned)f2bf(acc[mt][nt][2 * v + 1]) << 16);
        *(uint4*)&O[(((size_t)(b2 * NHEAD + h)) * DK + dk) * S_LEN + s0] =
            *(uint4*)pk;
      }
    }
}

// ---------------------------------------------------------------------------
// RoPE in-place on bf16 Q and K, layout [B,H,S,DK]; one thread per (bh,s,pair)
// ---------------------------------------------------------------------------
__global__ void rope_kernel(unsigned short* __restrict__ Q,
                            unsigned short* __restrict__ Kk,
                            const int* __restrict__ pos, int total) {
  int idx = blockIdx.x * blockDim.x + threadIdx.x;
  if (idx >= total) return;
  int i  = idx & 31;                 // pair index within head (freq index)
  int s  = (idx >> 5) & (S_LEN - 1);
  int bh = idx >> 16;                // 5 + 11 bits below
  float p = (float)pos[s];
  float freq = __powf(10000.0f, -(float)i * (1.0f / 32.0f));
  float ang = p * freq;
  float c, sn;
  __sincosf(ang, &sn, &c);
  size_t off = ((size_t)bh * S_LEN + s) * DK + 2 * i;

  unsigned* qp = (unsigned*)&Q[off];
  unsigned qv = *qp;
  float qe = bf2f((unsigned short)(qv & 0xFFFF));
  float qo = bf2f((unsigned short)(qv >> 16));
  *qp = (unsigned)f2bf(qe * c - qo * sn) |
        ((unsigned)f2bf(qo * c + qe * sn) << 16);

  unsigned* kp = (unsigned*)&Kk[off];
  unsigned kv = *kp;
  float ke = bf2f((unsigned short)(kv & 0xFFFF));
  float ko = bf2f((unsigned short)(kv >> 16));
  *kp = (unsigned)f2bf(ke * c - ko * sn) |
        ((unsigned)f2bf(ko * c + ke * sn) << 16);
}

// ---------------------------------------------------------------------------
// Flash attention. Grid (B*H, S/64), 128 threads (4 waves, 16 query rows each).
// Computes S^T = K.Q^T so each lane owns one query row's softmax state.
// Out^T = V^T.P^T accumulated in 4 WMMA tiles; epilogue packs bf16 [B,S,D].
// ---------------------------------------------------------------------------
__global__ __launch_bounds__(128)
void attn_kernel(const unsigned short* __restrict__ Q,
                 const unsigned short* __restrict__ K,
                 const unsigned short* __restrict__ Vt,
                 unsigned short* __restrict__ Oatt) {
  const int bh   = blockIdx.x;
  const int h    = bh & (NHEAD - 1);
  const int b    = bh >> 4;
  const int wave = threadIdx.x >> 5;
  const int lane = threadIdx.x & 31;
  const int lhalf = lane >> 4;
  const int l16   = lane & 15;
  const int qtile = blockIdx.y * 64 + wave * 16;
  const int iglob = qtile + l16;

  const unsigned short* qbase = Q  + ((size_t)bh * S_LEN + qtile) * DK;
  const unsigned short* kbase = K  + (size_t)bh * S_LEN * DK;
  const unsigned short* vbase = Vt + (size_t)bh * DK * S_LEN;

  // Q as B-operand: lane = query i, 16 contiguous halves at dk = c*32 + lhalf*16
  Frag qb[2];
  for (int c = 0; c < 2; c++) {
    const unsigned short* p = qbase + (size_t)l16 * DK + c * 32 + lhalf * 16;
    qb[c].q[0] = *(const uint4*)p;
    qb[c].q[1] = *(const uint4*)(p + 8);
  }

  v8f acc[4] = {};
  float mrow = -1e30f, lrow = 0.0f;
  const float scl = 0.125f;  // 1/sqrt(64)

  for (int j0 = 0; j0 < qtile + 16; j0 += 32) {
    v8f st[2];
    for (int jt = 0; jt < 2; jt++) {
      Frag ka[2];
      const unsigned short* kr = kbase + (size_t)(j0 + jt * 16 + l16) * DK;
      for (int c = 0; c < 2; c++) {
        ka[c].q[0] = *(const uint4*)(kr + c * 32 + lhalf * 16);
        ka[c].q[1] = *(const uint4*)(kr + c * 32 + lhalf * 16 + 8);
      }
      v8f z = {};
      z = __builtin_amdgcn_wmma_f32_16x16x32_bf16(false, ka[0].v, false, qb[0].v,
                                                  (short)0, z, false, false);
      z = __builtin_amdgcn_wmma_f32_16x16x32_bf16(false, ka[1].v, false, qb[1].v,
                                                  (short)0, z, false, false);
      st[jt] = z;
    }
    // scale + causal mask + row max (rows live per-lane: j in VGPRs)
    float rmax = -1e30f;
    for (int jt = 0; jt < 2; jt++)
      for (int v = 0; v < 8; v++) {
        int j = j0 + jt * 16 + v + lhalf * 8;
        float x = st[jt][v] * scl;
        x = (j <= iglob) ? x : -1e30f;
        st[jt][v] = x;
        rmax = fmaxf(rmax, x);
      }
    rmax = fmaxf(rmax, __shfl_xor(rmax, 16, 32));
    float mnew = fmaxf(mrow, rmax);
    float corr = __expf(mrow - mnew);
    float rsum = 0.0f;
    for (int v = 0; v < 8; v++) {
      float p0 = __expf(st[0][v] - mnew);
      float p1 = __expf(st[1][v] - mnew);
      st[0][v] = p0; st[1][v] = p1;
      rsum += p0 + p1;
    }
    rsum += __shfl_xor(rsum, 16, 32);
    mrow = mnew;
    lrow = lrow * corr + rsum;

    // Pack P into a bf16 B-operand fragment (K-dim = 32 keys)
    unsigned s0p[4], s1p[4];
    for (int v = 0; v < 4; v++) {
      s0p[v] = (unsigned)f2bf(st[0][2 * v]) | ((unsigned)f2bf(st[0][2 * v + 1]) << 16);
      s1p[v] = (unsigned)f2bf(st[1][2 * v]) | ((unsigned)f2bf(st[1][2 * v + 1]) << 16);
    }
    Frag pf;
    for (int v = 0; v < 4; v++) {
      unsigned x0 = __shfl_xor(s0p[v], 16, 32);
      unsigned x1 = __shfl_xor(s1p[v], 16, 32);
      pf.u[v]     = (lhalf == 0) ? s0p[v] : x1;   // K = 16*lhalf + 0..7
      pf.u[4 + v] = (lhalf == 0) ? x0 : s1p[v];   // K = 16*lhalf + 8..15
    }
    // rescale accumulators, then Out^T += V^T x P^T
    for (int d = 0; d < 4; d++)
      for (int v = 0; v < 8; v++) acc[d][v] *= corr;
    for (int d = 0; d < 4; d++) {
      Frag va;
      const unsigned short* vr = vbase + (size_t)(d * 16 + l16) * S_LEN + j0;
      va.q[0] = *(const uint4*)(vr + lhalf * 16);
      va.q[1] = *(const uint4*)(vr + lhalf * 16 + 8);
      acc[d] = __builtin_amdgcn_wmma_f32_16x16x32_bf16(
          false, va.v, false, pf.v, (short)0, acc[d], false, false);
    }
  }

  float rl = 1.0f / lrow;
  // Out^T tile d: lane = query i, VGPR v -> dk = d*16 + v + lhalf*8. Pack b128.
  for (int d = 0; d < 4; d++) {
    unsigned pk[4];
    for (int v = 0; v < 4; v++)
      pk[v] = (unsigned)f2bf(acc[d][2 * v] * rl) |
              ((unsigned)f2bf(acc[d][2 * v + 1] * rl) << 16);
    *(uint4*)&Oatt[((size_t)(b * S_LEN + iglob)) * DMODEL + h * DK + d * 16 +
                   lhalf * 8] = *(uint4*)pk;
  }
}

// ---------------------------------------------------------------------------
// Launch. Workspace usage: 88 MB total.
//   [0,16M)   x bf16        [16M,24M) wq/wk/wv/wo bf16
//   [24M,40M) Q bf16 [B,H,S,DK]   [40M,56M) K bf16 [B,H,S,DK]
//   [56M,72M) V^T bf16 [B,H,DK,S] [72M,88M) attn bf16 [B,S,D]
// ---------------------------------------------------------------------------
extern "C" void kernel_launch(void* const* d_in, const int* in_sizes, int n_in,
                              void* d_out, int out_size, void* d_ws,
                              size_t ws_size, hipStream_t stream) {
  const float* x  = (const float*)d_in[0];
  const float* wq = (const float*)d_in[1];
  const float* wk = (const float*)d_in[2];
  const float* wv = (const float*)d_in[3];
  const float* wo = (const float*)d_in[4];
  const int* tpos = (const int*)d_in[5];

  char* ws = (char*)d_ws;
  unsigned short* xb  = (unsigned short*)(ws);
  unsigned short* wqb = (unsigned short*)(ws + (16u << 20));
  unsigned short* wkb = wqb + (1u << 20);
  unsigned short* wvb = wkb + (1u << 20);
  unsigned short* wob = wvb + (1u << 20);
  unsigned short* qw  = (unsigned short*)(ws + (24u << 20));
  unsigned short* kw  = (unsigned short*)(ws + (40u << 20));
  unsigned short* vt  = (unsigned short*)(ws + (56u << 20));
  unsigned short* at  = (unsigned short*)(ws + (72u << 20));

  const int xpairs = BATCH * S_LEN * DMODEL / 2;  // 4,194,304
  const int wpairs = DMODEL * DMODEL / 2;         // 524,288
  cvt_f32_bf16<<<xpairs / 256, 256, 0, stream>>>(x, xb, xpairs);
  cvt_f32_bf16<<<wpairs / 256, 256, 0, stream>>>(wq, wqb, wpairs);
  cvt_f32_bf16<<<wpairs / 256, 256, 0, stream>>>(wk, wkb, wpairs);
  cvt_f32_bf16<<<wpairs / 256, 256, 0, stream>>>(wv, wvb, wpairs);
  cvt_f32_bf16<<<wpairs / 256, 256, 0, stream>>>(wo, wob, wpairs);

  dim3 gg(BATCH * S_LEN / 128, DMODEL / 128);  // (64, 8)
  gemm_bf16<<<gg, 256, 0, stream>>>(xb, wqb, qw, 0);
  gemm_bf16<<<gg, 256, 0, stream>>>(xb, wkb, kw, 0);
  gemm_bf16<<<gg, 256, 0, stream>>>(xb, wvb, vt, 1);

  const int rtot = BATCH * NHEAD * S_LEN * 32;  // 4,194,304
  rope_kernel<<<rtot / 256, 256, 0, stream>>>(qw, kw, tpos, rtot);

  attn_kernel<<<dim3(BATCH * NHEAD, S_LEN / 64), 128, 0, stream>>>(qw, kw, vt, at);

  gemm_bf16<<<gg, 256, 0, stream>>>(at, wob, d_out, 2);
}
